// Attention_46875273068859
// MI455X (gfx1250) — matrix operand order repaired and checked by reading, not compiled, not run on hardware
//
#include <hip/hip_runtime.h>

// ---------------------------------------------------------------------------
// MI455X (gfx1250) fused attention block:
//   qkv = x @ Wqkv + b ; per-head flash attention ; out = attn @ Wfc + b
// bf16 WMMA (f32 accum) everywhere; GEMMs are LDS-staged with double-buffered
// Tensor-Data-Mover (tensor_load_to_lds) feeds; flash attention streams K via
// TDM and keeps softmax entirely in registers/LDS.
// ---------------------------------------------------------------------------

typedef __attribute__((ext_vector_type(16))) __bf16         v16bf;
typedef __attribute__((ext_vector_type(16))) unsigned short v16u;
typedef __attribute__((ext_vector_type(8)))  unsigned short u16x8;
typedef __attribute__((ext_vector_type(8)))  float          v8f;
typedef __attribute__((ext_vector_type(4)))  unsigned int   u32x4;
typedef __attribute__((ext_vector_type(8)))  int            i32x8;

#define DIM    1024
#define NHEAD  16
#define HDIM   64
#define SEQ    4096
#define BATCH  2
#define MROWS  (BATCH * SEQ)      /* 8192 */
#define ATT_SCALE 0.125f          /* 64^-0.5 */

#define WMMA_BF16(a, b, c) \
  __builtin_amdgcn_wmma_f32_16x16x32_bf16(false, (a), false, (b), (short)0, (c), false, false)

union FragAB {
  v16u  u;
  u16x8 h[2];
  v16bf b;
};

__device__ __forceinline__ unsigned short f2bf(float f) {
  unsigned int u = __float_as_uint(f);
  u += 0x7FFFu + ((u >> 16) & 1u);      // round-to-nearest-even
  return (unsigned short)(u >> 16);
}

// ---------------------------------------------------------------------------
// Tensor Data Mover: 2D tile (bf16) global -> LDS, with optional LDS row
// padding (D# pad_enable) for bank-conflict-free fragment reads.
// D# group layout per CDNA5 ISA; <=2D tensors
// use the two-SGPR-group form (VADDR2/3 = NULL).
__device__ __forceinline__ void tdm_load_2d(unsigned ldsOff,
                                            const unsigned short* gptr,
                                            unsigned tileW,        // elems
                                            unsigned tileH,        // rows
                                            unsigned strideElems,  // row stride
                                            unsigned padIntCode,   // 2^c * 8B
                                            unsigned padAmtCode) { // (c+1) DW
  unsigned long long ga = (unsigned long long)(const void*)gptr;
  u32x4 g0;
  g0[0] = 1u;                                              // count=1, user D#
  g0[1] = ldsOff;                                          // lds_addr (bytes)
  g0[2] = (unsigned)(ga & 0xFFFFFFFFull);                  // global_addr lo
  g0[3] = (unsigned)((ga >> 32) & 0x1FFFFFFull) | (2u << 30); // hi | type=2

  unsigned d0 = (1u << 16);                                // data_size = 2B
  if (padAmtCode | padIntCode)
    d0 |= (1u << 20) | (padIntCode << 22) | (padAmtCode << 25);
  i32x8 g1;
  g1[0] = (int)d0;
  g1[1] = (int)((strideElems & 0xFFFFu) << 16);            // tensor_dim0 lo
  g1[2] = (int)(((strideElems >> 16) & 0xFFFFu) |          // tensor_dim0 hi
                ((tileH & 0xFFFFu) << 16));                // tensor_dim1 lo
  g1[3] = (int)(((tileH >> 16) & 0xFFFFu) |                // tensor_dim1 hi
                ((tileW & 0xFFFFu) << 16));                // tile_dim0
  g1[4] = (int)(tileH & 0xFFFFu);                          // tile_dim1 (dim2=0)
  g1[5] = (int)strideElems;                                // dim0_stride lo
  g1[6] = 0;                                               // dim0_stride hi
  g1[7] = 0;
  asm volatile("tensor_load_to_lds %0, %1" :: "s"(g0), "s"(g1) : "memory");
}

__device__ __forceinline__ void tdm_wait0() {
  asm volatile("s_wait_tensorcnt 0x0" ::: "memory");
}

// ---------------------------------------------------------------------------
// fp32 -> bf16 elementwise
__global__ __launch_bounds__(256)
void cvt_bf16(const float* __restrict__ in, unsigned short* __restrict__ out, int n) {
  int i = blockIdx.x * 256 + threadIdx.x;
  if (i < n) out[i] = f2bf(in[i]);
}

// fp32 W[k][n] -> bf16 Wt[n][k]  (so GEMM B-fragments read contiguous K)
__global__ __launch_bounds__(256)
void cvt_bf16_t(const float* __restrict__ in, unsigned short* __restrict__ out,
                int K, int N) {
  int i = blockIdx.x * 256 + threadIdx.x;
  if (i < K * N) {
    int n = i / K, k = i - n * K;
    out[i] = f2bf(in[k * N + n]);
  }
}

// ---------------------------------------------------------------------------
// LDS-staged GEMM core: block = 8 waves, block tile 256(M) x 64(N), K-chunk 32,
// TDM double-buffered. Each wave computes 32x64 of C (8 WMMA tiles / chunk).
// LDS rows padded 32 -> 40 elems (TDM pad: every 64B insert 16B).
#define LROW 40
#define GEMM_CORE(A_, Bt_)                                                     \
  const int lane = threadIdx.x & 31;                                           \
  const int wave = threadIdx.x >> 5;                                           \
  const int half = lane >> 4;                                                  \
  const int l16  = lane & 15;                                                  \
  const int kbA  = half * 8;                                                   \
  const int kbB  = half * 16;                                                  \
  const int mBase = blockIdx.y * 256;                                          \
  const int colBase = blockIdx.x * 64;                                         \
  __shared__ unsigned short sA[2][256 * LROW];                                 \
  __shared__ unsigned short sB[2][64 * LROW];                                  \
  v8f c[2][4];                                                                 \
  _Pragma("unroll") for (int st = 0; st < 2; ++st)                             \
  _Pragma("unroll") for (int ct = 0; ct < 4; ++ct)                             \
  _Pragma("unroll") for (int j = 0; j < 8; ++j) c[st][ct][j] = 0.f;            \
  if (wave == 0) {                                                             \
    tdm_load_2d((unsigned)(size_t)&sA[0][0], A_ + (size_t)mBase * DIM,         \
                32, 256, DIM, 3, 3);                                           \
    tdm_load_2d((unsigned)(size_t)&sB[0][0], Bt_ + (size_t)colBase * DIM,      \
                32, 64, DIM, 3, 3);                                            \
    tdm_wait0();                                                               \
  }                                                                            \
  __syncthreads();                                                             \
  for (int ic = 0; ic < DIM / 32; ++ic) {                                      \
    const int cur = ic & 1;                                                    \
    if (ic + 1 < DIM / 32 && wave == 0) {                                      \
      const int k0 = (ic + 1) * 32;                                            \
      tdm_load_2d((unsigned)(size_t)&sA[cur ^ 1][0],                           \
                  A_ + (size_t)mBase * DIM + k0, 32, 256, DIM, 3, 3);          \
      tdm_load_2d((unsigned)(size_t)&sB[cur ^ 1][0],                           \
                  Bt_ + (size_t)colBase * DIM + k0, 32, 64, DIM, 3, 3);        \
    }                                                                          \
    FragAB aF[2], bF[4];                                                       \
    _Pragma("unroll") for (int st = 0; st < 2; ++st) {                         \
      const unsigned short* p = &sA[cur][(wave * 32 + st * 16 + l16) * LROW];  \
      aF[st].h[0] = *(const u16x8*)(p + kbA);                                  \
      aF[st].h[1] = *(const u16x8*)(p + kbA + 16);                             \
    }                                                                          \
    _Pragma("unroll") for (int ct = 0; ct < 4; ++ct)                           \
    {                                                                          \
      bF[ct].h[0] = *(const u16x8*)(&sB[cur][(ct * 16 + l16) * LROW] + kbA);   \
      bF[ct].h[1] = *(const u16x8*)(&sB[cur][(ct * 16 + l16) * LROW] + kbA + 16); \
    }                                                                          \
    _Pragma("unroll") for (int st = 0; st < 2; ++st)                           \
    _Pragma("unroll") for (int ct = 0; ct < 4; ++ct)                           \
      c[st][ct] = WMMA_BF16(aF[st].b, bF[ct].b, c[st][ct]);                    \
    if (wave == 0) tdm_wait0();                                                \
    __syncthreads();                                                           \
  }

// QKV GEMM: C[8192][3072] = A * Wqkv + bias, scattered to q/k/v[b][h][n][64].
__global__ __launch_bounds__(256)
void gemm_qkv(const unsigned short* __restrict__ A,
              const unsigned short* __restrict__ Bt,   // [3072][1024]
              const float*          __restrict__ bias, // [3072]
              unsigned short*       __restrict__ qkv) {
  GEMM_CORE(A, Bt)
#pragma unroll
  for (int ct = 0; ct < 4; ++ct) {
    const int col = colBase + ct * 16 + l16;          // [0, 3072)
    const float bv = bias[col];
    const int sel = col >> 10;                        // 0=q 1=k 2=v
    const int rem = col & 1023;
    const int hh  = rem >> 6;
    const int dd  = rem & 63;
#pragma unroll
    for (int st = 0; st < 2; ++st)
#pragma unroll
      for (int j = 0; j < 8; ++j) {
        const int row = mBase + wave * 32 + st * 16 + half * 8 + j;
        const int bb  = row >> 12;
        const int nn  = row & 4095;
        size_t addr = (size_t)sel * (MROWS * DIM)
                    + ((size_t)(bb * NHEAD + hh) * SEQ + nn) * HDIM + dd;
        qkv[addr] = f2bf(c[st][ct][j] + bv);
      }
  }
}

// FC GEMM: out[8192][1024] = A * Wfc + bias, fp32 output.
__global__ __launch_bounds__(256)
void gemm_fc(const unsigned short* __restrict__ A,
             const unsigned short* __restrict__ Bt,   // [1024][1024]
             const float*          __restrict__ bias, // [1024]
             float*                __restrict__ out) {
  GEMM_CORE(A, Bt)
#pragma unroll
  for (int ct = 0; ct < 4; ++ct) {
    const int col = colBase + ct * 16 + l16;
    const float bv = bias[col];
#pragma unroll
    for (int st = 0; st < 2; ++st)
#pragma unroll
      for (int j = 0; j < 8; ++j) {
        const int row = mBase + wave * 32 + st * 16 + half * 8 + j;
        out[(size_t)row * DIM + col] = c[st][ct][j] + bv;
      }
  }
}

// ---------------------------------------------------------------------------
// Flash attention: block = 4 waves x 16 query rows, one (batch,head).
// K chunk arrives via TDM (padded rows), V transposed cooperatively,
// online softmax in registers, P restaged through LDS for the PV WMMAs.
#define KROW 72   /* 64 elems + 16B TDM pad  (144B rows) */
#define VROW 40   /* 32 elems + manual pad   (80B rows)  */
__global__ __launch_bounds__(128)
void flash_attn(const unsigned short* __restrict__ Q,   // [bh][n][64]
                const unsigned short* __restrict__ K,
                const unsigned short* __restrict__ V,
                unsigned short*       __restrict__ O) { // [b][n][1024]
  __shared__ unsigned short sK[32 * KROW];
  __shared__ unsigned short sVt[64 * VROW];
  __shared__ unsigned short sP[4][16 * VROW];

  const int tid  = threadIdx.x;
  const int lane = tid & 31;
  const int wave = tid >> 5;
  const int half = lane >> 4;
  const int l16  = lane & 15;
  const int kbA  = half * 8;
  const int kbB  = half * 16;

  const int bh = blockIdx.y;                    // [0, 32)
  const size_t base = (size_t)bh * SEQ * HDIM;
  const int qRow = blockIdx.x * 64 + wave * 16;

  // Q A-fragments held for the whole key sweep: d-chunks [0,32) and [32,64)
  const unsigned short* qPtr = Q + base + (size_t)(qRow + l16) * HDIM;
  FragAB qa0, qa1;
  qa0.h[0] = *(const u16x8*)(qPtr + kbA);
  qa0.h[1] = *(const u16x8*)(qPtr + kbA + 16);
  qa1.h[0] = *(const u16x8*)(qPtr + 32 + kbA);
  qa1.h[1] = *(const u16x8*)(qPtr + 32 + kbA + 16);

  v8f acc[4];
  float mRun[8], lRun[8];
#pragma unroll
  for (int dt = 0; dt < 4; ++dt)
#pragma unroll
    for (int j = 0; j < 8; ++j) acc[dt][j] = 0.f;
#pragma unroll
  for (int j = 0; j < 8; ++j) { mRun[j] = -3.0e38f; lRun[j] = 0.f; }

  // cooperative V mapping: 128 threads x 16 bf16 = one 32x64 tile
  const int cr = tid >> 2;            // key row in chunk [0,32)
  const int cc = (tid & 3) * 16;      // d offset {0,16,32,48}

  for (int kc = 0; kc < SEQ; kc += 32) {
    __syncthreads();
    if (wave == 0)
      tdm_load_2d((unsigned)(size_t)&sK[0], K + base + (size_t)kc * HDIM,
                  64, 32, 64, 4, 3);
    {
      const size_t g = base + (size_t)(kc + cr) * HDIM + cc;
      v16u vv = *(const v16u*)(V + g);
#pragma unroll
      for (int j = 0; j < 16; ++j) sVt[(cc + j) * VROW + cr] = vv[j];
      if (kc + 32 < SEQ) __builtin_prefetch(V + g + 32 * HDIM, 0, 0);
    }
    if (wave == 0) tdm_wait0();
    __syncthreads();

    // S = (Q K^T) * scale for two 16-key subtiles  (K-dim = D = 64)
    v8f s[2];
#pragma unroll
    for (int st = 0; st < 2; ++st) {
      const unsigned short* kRow = sK + (st * 16 + l16) * KROW + kbA;
      FragAB b0, b1;
      b0.h[0] = *(const u16x8*)(kRow);
      b0.h[1] = *(const u16x8*)(kRow + 16);
      b1.h[0] = *(const u16x8*)(kRow + 32);
      b1.h[1] = *(const u16x8*)(kRow + 48);
      v8f z;
#pragma unroll
      for (int j = 0; j < 8; ++j) z[j] = 0.f;
      z = WMMA_BF16(qa0.b, b0.b, z);
      z = WMMA_BF16(qa1.b, b1.b, z);
#pragma unroll
      for (int j = 0; j < 8; ++j) z[j] *= ATT_SCALE;
      s[st] = z;
    }

    // online softmax: row stats via 16-lane xor-shuffles
    float tmax[8];
#pragma unroll
    for (int j = 0; j < 8; ++j) tmax[j] = fmaxf(s[0][j], s[1][j]);
#pragma unroll
    for (int m = 1; m < 16; m <<= 1)
#pragma unroll
      for (int j = 0; j < 8; ++j) tmax[j] = fmaxf(tmax[j], __shfl_xor(tmax[j], m));

    float corr[8], tsum[8];
#pragma unroll
    for (int j = 0; j < 8; ++j) {
      const float mN = fmaxf(mRun[j], tmax[j]);
      corr[j] = __expf(mRun[j] - mN);
      mRun[j] = mN;
      const float p0 = __expf(s[0][j] - mN);
      const float p1 = __expf(s[1][j] - mN);
      s[0][j] = p0; s[1][j] = p1;
      tsum[j] = p0 + p1;
    }
#pragma unroll
    for (int m = 1; m < 16; m <<= 1)
#pragma unroll
      for (int j = 0; j < 8; ++j) tsum[j] += __shfl_xor(tsum[j], m);
#pragma unroll
    for (int j = 0; j < 8; ++j) lRun[j] = lRun[j] * corr[j] + tsum[j];
#pragma unroll
    for (int dt = 0; dt < 4; ++dt)
#pragma unroll
      for (int j = 0; j < 8; ++j) acc[dt][j] *= corr[j];

    // stage P (C-layout -> A-layout) through wave-private LDS
    unsigned short* myP = sP[wave];
#pragma unroll
    for (int j = 0; j < 8; ++j) {
      const int r = half * 8 + j;
      myP[r * VROW + l16]      = f2bf(s[0][j]);
      myP[r * VROW + 16 + l16] = f2bf(s[1][j]);
    }
    FragAB pa;
    const unsigned short* pRow = myP + l16 * VROW + kbA;
    pa.h[0] = *(const u16x8*)(pRow);
    pa.h[1] = *(const u16x8*)(pRow + 16);

    // O += P * V   (K-dim = 32 keys, 4 d-tiles of 16)
#pragma unroll
    for (int dt = 0; dt < 4; ++dt) {
      FragAB vb;
      vb.h[0] = *(const u16x8*)(sVt + (dt * 16 + l16) * VROW + kbA);
      vb.h[1] = *(const u16x8*)(sVt + (dt * 16 + l16) * VROW + kbA + 16);
      acc[dt] = WMMA_BF16(pa.b, vb.b, acc[dt]);
    }
  }

  // epilogue: normalize by row sum, write bf16 into [b][n][h*64+d]
  const int bb = bh >> 4;
  const int hd = bh & 15;
#pragma unroll
  for (int dt = 0; dt < 4; ++dt) {
#pragma unroll
    for (int j = 0; j < 8; ++j) {
      const int n = qRow + half * 8 + j;
      const int d = dt * 16 + l16;
      const float val = acc[dt][j] / lRun[j];
      O[((size_t)(bb * SEQ + n)) * DIM + hd * HDIM + d] = f2bf(val);
    }
  }
}

// ---------------------------------------------------------------------------
extern "C" void kernel_launch(void* const* d_in, const int* in_sizes, int n_in,
                              void* d_out, int out_size, void* d_ws, size_t ws_size,
                              hipStream_t stream) {
  const float* x    = (const float*)d_in[0];
  const float* Wqkv = (const float*)d_in[1];
  const float* bqkv = (const float*)d_in[2];
  const float* Wfc  = (const float*)d_in[3];
  const float* bfc  = (const float*)d_in[4];
  float* out = (float*)d_out;

  // workspace layout (bytes), total ~92.3 MB
  char* ws = (char*)d_ws;
  unsigned short* xbf   = (unsigned short*)(ws);                       // 16 MB
  unsigned short* wqkvt = (unsigned short*)(ws + 16777216);            //  6 MB
  unsigned short* wfct  = (unsigned short*)(ws + 16777216 + 6291456);  //  2 MB
  unsigned short* qkv   = (unsigned short*)(ws + 25165824);            // 48 MB
  unsigned short* attn  = (unsigned short*)(ws + 75497472);            // 16 MB

  cvt_bf16  <<<(MROWS * DIM) / 256, 256, 0, stream>>>(x, xbf, MROWS * DIM);
  cvt_bf16_t<<<(DIM * 3 * DIM) / 256, 256, 0, stream>>>(Wqkv, wqkvt, DIM, 3 * DIM);
  cvt_bf16_t<<<(DIM * DIM) / 256, 256, 0, stream>>>(Wfc, wfct, DIM, DIM);

  gemm_qkv<<<dim3(48, 32), 256, 0, stream>>>(xbf, wqkvt, bqkv, qkv);

  flash_attn<<<dim3(SEQ / 64, BATCH * NHEAD), 128, 0, stream>>>(
      qkv, qkv + (size_t)MROWS * DIM, qkv + 2 * (size_t)MROWS * DIM, attn);

  gemm_fc<<<dim3(16, 32), 256, 0, stream>>>(attn, wfct, bfc, out);
}
